// VectorKuramoto_7928509628758
// MI455X (gfx1250) — matrix operand order repaired and checked by reading, not compiled, not run on hardware
//
#include <hip/hip_runtime.h>

typedef __attribute__((ext_vector_type(16))) _Float16 v16h;
typedef __attribute__((ext_vector_type(8)))  float    v8f;
typedef __attribute__((ext_vector_type(2)))  __fp16   h2;
typedef __attribute__((ext_vector_type(8)))  unsigned v8u;

#define NB 4
#define NN 12288
#define ND 4
#define NC 4096
#define INV2PI 0.15915494309189535f

__global__ __launch_bounds__(256) void vk_kuramoto_wmma(
    const float* __restrict__ theta,
    const float* __restrict__ gammap,
    const float* __restrict__ spikes,
    float* __restrict__ outp)
{
    __shared__ float    s_spk[NC];        // 16 KB: group spikes (f32)
    __shared__ unsigned s_sc[ND][NC];     // 64 KB: packed {sin f16 | cos f16} of theta
    __shared__ float    s_red[16];
    __shared__ float    s_invrng;

    const int g   = blockIdx.x >> 5;      // group 0..11
    const int c   = g >> 2;               // channel 0..2
    const int b   = g & 3;                // batch 0..3
    const int tid = threadIdx.x;
    const int wave = tid >> 5;
    const int lane = tid & 31;

    // ---------------- prologue: gather + sincos table + rng reduction ----------------
    float lmin = 3.4e38f, lmax = -3.4e38f;
    for (int j = tid; j < NC; j += 256) {
        const int node = c + 3 * j;
        const float sv = spikes[b * NN + node];
        s_spk[j] = sv;
        lmin = fminf(lmin, sv);
        lmax = fmaxf(lmax, sv);
        const float4 tv = *(const float4*)(theta + ((size_t)(b * NN + node)) * ND);
        const float td[4] = {tv.x, tv.y, tv.z, tv.w};
#pragma unroll
        for (int d = 0; d < 4; ++d) {
            const float t  = td[d] * INV2PI;
            const float sn = __builtin_amdgcn_sinf(t);
            const float cs = __builtin_amdgcn_cosf(t);
            h2 pk = __builtin_amdgcn_cvt_pkrtz(sn, cs);   // sin lo | cos hi
            s_sc[d][j] = __builtin_bit_cast(unsigned, pk);
        }
    }
#pragma unroll
    for (int m = 16; m >= 1; m >>= 1) {
        lmin = fminf(lmin, __shfl_xor(lmin, m, 32));
        lmax = fmaxf(lmax, __shfl_xor(lmax, m, 32));
    }
    if (lane == 0) { s_red[wave] = lmin; s_red[8 + wave] = lmax; }
    __syncthreads();
    if (tid == 0) {
        float mn = s_red[0], mx = s_red[8];
        for (int w = 1; w < 8; ++w) { mn = fminf(mn, s_red[w]); mx = fmaxf(mx, s_red[8 + w]); }
        s_invrng = 1.0f / fmaxf(mx - mn, 1e-6f);
    }
    __syncthreads();
    const float invrng = s_invrng;

    // ---------------- per-wave tile setup ----------------
    const int half  = lane >> 4;                       // 0/1
    const int l15   = lane & 15;
    const int tile  = ((blockIdx.x & 31) << 3) + wave; // 0..255
    const int ibase = tile << 4;                       // receiver base (group-local)
    const float s_i = s_spk[ibase + l15];              // A row spike
    const int jbA   = half ? 4 : 0;                    // A lane j offsets: jbA+{0..3}, jbA+8+{0..3}
    const int jbB   = half ? 4 : 0;                    // B lane j offsets: jbB+{0..3}, jbB+8+{0..3}
    const int dcol  = l15 & 3;                         // B column dim
    const int blk   = l15 >> 2;                        // column block 0..3 (S1,S2,S3,S4)
    const unsigned csShift = (blk & 1) ? 16u : 0u;     // blk 1,3 -> cos (hi16)
    const bool oddSlot = (blk >= 2);                   // blk 2,3 -> Q rows (odd K)

    v8f acc = {};

    // ---------------- main loop: 256 iterations of 16 senders ----------------
    for (int k = 0; k < 256; ++k) {
        const int kb = k << 4;

        // A: P/Q for 8 (i,j) pairs, interleaved K = 2j(P), 2j+1(Q)
        const float4 sa = *(const float4*)&s_spk[kb + jbA];
        const float4 sb = *(const float4*)&s_spk[kb + jbA + 8];
        const float sj[8] = {sa.x, sa.y, sa.z, sa.w, sb.x, sb.y, sb.z, sb.w};
        v8u aw;
#pragma unroll
        for (int t = 0; t < 8; ++t) {
            const float nd    = fabsf(s_i - sj[t]) * invrng;
            const float alpha = 0.25f * nd;
            const float aff   = 0.25f - alpha;
            const float tt    = alpha * INV2PI;
            const float sA    = __builtin_amdgcn_sinf(tt);
            const float cA    = __builtin_amdgcn_cosf(tt);
            h2 pk = __builtin_amdgcn_cvt_pkrtz(aff * cA, aff * sA); // P lo, Q hi
            aw[t] = __builtin_bit_cast(unsigned, pk);
        }

        // B: column dcol of [sin|cos|sin|cos] blocks, zeros in the other K parity
        const unsigned* scrow = &s_sc[dcol][kb + jbB];
        const uint4 w0 = *(const uint4*)scrow;
        const uint4 w1 = *(const uint4*)(scrow + 8);
        const unsigned wv[8] = {w0.x, w0.y, w0.z, w0.w, w1.x, w1.y, w1.z, w1.w};
        v8u bw;
#pragma unroll
        for (int t = 0; t < 8; ++t) {
            const unsigned hv = (wv[t] >> csShift) & 0xffffu;
            bw[t] = oddSlot ? (hv << 16) : hv;
        }

        const v16h Av = __builtin_bit_cast(v16h, aw);
        const v16h Bv = __builtin_bit_cast(v16h, bw);
        acc = __builtin_amdgcn_wmma_f32_16x16x32_f16(
            false, Av, false, Bv, (short)0, acc, false, false);
    }

    // ---------------- epilogue: combine S1..S4, add gamma, normalize ----------------
    const float inv_n = 1.0f / (float)NC;
#pragma unroll
    for (int r = 0; r < 8; ++r) {
        const float x = acc[r];
        const float y = __shfl_xor(x, 12, 32);          // blk0<->3, blk1<->2
        const float sx = (blk == 3) ? -x : x;
        const float sy = (blk == 0) ? -y : y;
        const float z = sx + sy;                        // blk0/3: U=S1-S4, blk1/2: V=S2+S3
        const float w = __shfl_xor(z, 4, 32);           // blk0<->1, blk2<->3
        const bool  uIsZ = (blk == 0) || (blk == 3);
        const float U = uIsZ ? z : w;
        const float V = uIsZ ? w : z;

        const int i    = ibase + (half << 3) + r;       // receiver (group-local)
        const int node = c + 3 * i;
        const size_t base = ((size_t)(b * NN + node)) * ND + dcol;

        const float th = theta[base];
        const float tt = th * INV2PI;
        const float si = __builtin_amdgcn_sinf(tt);
        const float ci = __builtin_amdgcn_cosf(tt);
        const float cpl = (ci * U - si * V) * inv_n;
        const float tn  = gammap[base] + cpl;           // theta + (gamma-theta) + coupling

        float ss = tn * tn;                             // norm over D within a 4-lane quad
        ss += __shfl_xor(ss, 1, 32);
        ss += __shfl_xor(ss, 2, 32);
        const float nrm = fmaxf(sqrtf(ss), 1e-6f);
        const float val = tn / nrm;
        if (blk == 0) outp[base] = val;                 // blks are redundant copies
    }
}

extern "C" void kernel_launch(void* const* d_in, const int* in_sizes, int n_in,
                              void* d_out, int out_size, void* d_ws, size_t ws_size,
                              hipStream_t stream) {
    (void)in_sizes; (void)n_in; (void)d_ws; (void)ws_size; (void)out_size;
    const float* theta  = (const float*)d_in[0];
    const float* gammap = (const float*)d_in[1];
    const float* aff    = (const float*)d_in[2];
    float* outp = (float*)d_out;
    // 12 groups * 32 blocks/group; 256 threads = 8 waves, 1 receiver-tile per wave
    vk_kuramoto_wmma<<<dim3(384), dim3(256), 0, stream>>>(theta, gammap, aff, outp);
}
